// MultiHeadSelfAttention_78073915507261
// MI455X (gfx1250) — compile-verified, matrix operand order checked by reading, not run on hardware
//
#include <hip/hip_runtime.h>
#include <hip/hip_bf16.h>

// ---------- problem constants ----------
#define BATCH 2
#define SEQ   2048
#define CDIM  1024
#define NHEAD 16
#define HDIM  64
#define BT    (BATCH * SEQ)     // 4096
#define C3    (3 * CDIM)        // 3072

typedef __attribute__((ext_vector_type(16))) _Float16 v16h;
typedef __attribute__((ext_vector_type(8)))  float    v8f;

union HV { unsigned int u[8]; v16h v; };

static __device__ __forceinline__ unsigned short f2h(float f) {
    _Float16 h = (_Float16)f;
    union { _Float16 h; unsigned short u; } cv; cv.h = h; return cv.u;
}

// A-layout k-index for 16-bit A matrix 16x32 (ISA 7.12.2):
// dword v (0..7), lane-group g (0/1), low half -> k; high half -> k+1
static __device__ __forceinline__ int apat(int v, int g) {
    return ((v & 4) << 2) + (g << 3) + ((v & 3) << 1);
}

// async global -> LDS copy, 16B per lane; tracked by ASYNCcnt
static __device__ __forceinline__ void async_b128(unsigned lds_off, const void* gaddr) {
    asm volatile("global_load_async_to_lds_b128 %0, %1, off"
                 :: "v"(lds_off), "v"(gaddr)
                 : "memory");
}
static __device__ __forceinline__ void wait_async0() {
    asm volatile("s_wait_asynccnt 0x0" ::: "memory");
}

// ---------- stage 0: f32 -> f16 conversion ----------
__global__ __launch_bounds__(256) void cvt_f32_f16(const float* __restrict__ src,
                                                   unsigned short* __restrict__ dst, int n) {
    int i = blockIdx.x * blockDim.x + threadIdx.x;
    int stride = gridDim.x * blockDim.x;
    for (; i < n; i += stride) dst[i] = f2h(src[i]);
}

// ---------- stage 1: QKV projection ----------
// qkv[t][j] = sum_c x[t][c] * W[j][c] + b[j]
// wave = one 16x64 tile of (BT x 3C).  Writes f16 Q,K as [b,h,t,d] and V as [b,h,d,t].
__global__ __launch_bounds__(256) void qkv_gemm(const unsigned short* __restrict__ xh,
                                                const unsigned short* __restrict__ wh,
                                                const float* __restrict__ bias,
                                                unsigned short* __restrict__ qh,
                                                unsigned short* __restrict__ kh,
                                                unsigned short* __restrict__ vth) {
    const int wave = (blockIdx.x * blockDim.x + threadIdx.x) >> 5;
    const int lane = threadIdx.x & 31;
    const int g    = lane >> 4;
    const int n16  = lane & 15;

    const int tileM = wave % (BT / 16);   // 0..255
    const int tileN = wave / (BT / 16);   // 0..47
    const int t0 = tileM * 16;
    const int j0 = tileN * 64;

    v8f acc0 = {}, acc1 = {}, acc2 = {}, acc3 = {};

    const unsigned short* xrowBase = xh + (size_t)(t0 + n16) * CDIM;

    for (int k0 = 0; k0 < CDIM; k0 += 32) {
        HV a;
#pragma unroll
        for (int v = 0; v < 8; ++v)
            a.u[v] = *(const unsigned int*)(xrowBase + k0 + apat(v, g));

#pragma unroll
        for (int aidx = 0; aidx < 4; ++aidx) {
            const unsigned short* wrow = wh + (size_t)(j0 + 16 * aidx + n16) * CDIM + k0;
            HV b;
#pragma unroll
            for (int v = 0; v < 8; ++v)
                b.u[v] = *(const unsigned int*)(wrow + apat(v, g));
            v8f* accp = (aidx == 0) ? &acc0 : (aidx == 1) ? &acc1 : (aidx == 2) ? &acc2 : &acc3;
            *accp = __builtin_amdgcn_wmma_f32_16x16x32_f16(false, a.v, false, b.v,
                                                           (short)0, *accp, false, false);
        }
    }

    // epilogue: bias add, route to Q (x32 scale) / K / V^T
    const int region = j0 / CDIM;            // 0=Q, 1=K, 2=V (uniform per wave)
    const int h = (j0 % CDIM) / HDIM;        // uniform per wave (j0 is 64-aligned)
    const int b = t0 / SEQ;

#pragma unroll
    for (int aidx = 0; aidx < 4; ++aidx) {
        const int j = j0 + 16 * aidx + n16;
        const int d = 16 * aidx + n16;       // 0..63
        const float bj = bias[j];
        const v8f acc = (aidx == 0) ? acc0 : (aidx == 1) ? acc1 : (aidx == 2) ? acc2 : acc3;
#pragma unroll
        for (int r = 0; r < 8; ++r) {
            const int m  = r + (g << 3);
            const int tt = (t0 + m) % SEQ;
            const float val = acc[r] + bj;
            if (region == 0) {
                qh[(((size_t)(b * NHEAD + h) * SEQ + tt) * HDIM) + d] = f2h(val * 32.0f);
            } else if (region == 1) {
                kh[(((size_t)(b * NHEAD + h) * SEQ + tt) * HDIM) + d] = f2h(val);
            } else {
                vth[(((size_t)(b * NHEAD + h) * HDIM + d) * SEQ) + tt] = f2h(val);
            }
        }
    }
}

// ---------- stage 2: flash attention ----------
// block = 4 waves = 64 query rows of one (b,h).  K/V tiles are async-copied into
// double-buffered LDS once per block (shared by all 4 waves); online softmax per wave.
__global__ __launch_bounds__(128) void flash_attn(const unsigned short* __restrict__ qh,
                                                  const unsigned short* __restrict__ kh,
                                                  const unsigned short* __restrict__ vth,
                                                  const unsigned char* __restrict__ mask,
                                                  float* __restrict__ out) {
    __shared__ alignas(16) unsigned short kbuf[2][32 * HDIM];  // 32 keys x 64 d   (4KB each)
    __shared__ alignas(16) unsigned short vbuf[2][HDIM * 32];  // 64 d    x 32 keys(4KB each)
    __shared__ unsigned short ldsP[4 * 16 * 32];               // per-wave P tiles

    const int tid  = threadIdx.x;
    const int wid  = tid >> 5;
    const int lane = tid & 31;
    const int g    = lane >> 4;
    const int n16  = lane & 15;

    const int qchunk = blockIdx.x % (SEQ / 64);   // 0..31
    const int bh     = blockIdx.x / (SEQ / 64);   // 0..31
    const int b      = bh / NHEAD;
    const int h      = bh % NHEAD;
    const int q0b    = qchunk * 64;               // block's first query row
    const int q0     = q0b + wid * 16;            // this wave's first query row
    const int kend   = q0b + 64;                  // block-uniform causal key limit

    const unsigned short* qbase = qh  + (size_t)bh * SEQ * HDIM;
    const unsigned short* kbase = kh  + (size_t)bh * SEQ * HDIM;
    const unsigned short* vbase = vth + (size_t)bh * HDIM * SEQ;
    const unsigned char*  mbase = mask + (size_t)b * SEQ * SEQ;
    volatile unsigned short* myP = ldsP + wid * 16 * 32;

    // issue async copy of K/V tiles for key block `kb` into buffer `bsel`
    auto issue_tiles = [&](int bsel, int kb) {
        // K tile: keys kb..kb+31 are a contiguous 4KB row-major chunk
        const char* kg = (const char*)(kbase + (size_t)kb * HDIM);
        unsigned    kl = (unsigned)(size_t)&kbuf[bsel][0];
        async_b128(kl + tid * 16,        kg + tid * 16);
        async_b128(kl + 2048 + tid * 16, kg + 2048 + tid * 16);
        // V tile from V^T: 64 d-rows, 64 bytes each (stride SEQ*2 in global)
        const int dd = tid >> 1, part = tid & 1;
        const char* vg = (const char*)(vbase + (size_t)dd * SEQ + kb) + part * 16;
        unsigned    vl = (unsigned)(size_t)&vbuf[bsel][0] + dd * 64 + part * 16;
        async_b128(vl,      vg);
        async_b128(vl + 32, vg + 32);
    };

    issue_tiles(0, 0);   // prefetch first key block before Q loads

    // load Q tile (16 x 64) in A-layout, held for the whole key loop
    v16h qa[2];
    {
        const unsigned short* qrow = qbase + (size_t)(q0 + n16) * HDIM;
#pragma unroll
        for (int c = 0; c < 2; ++c) {
            HV u;
#pragma unroll
            for (int v = 0; v < 8; ++v)
                u.u[v] = *(const unsigned int*)(qrow + c * 32 + apat(v, g));
            qa[c] = u.v;
        }
    }

    v8f o0 = {}, o1 = {}, o2 = {}, o3 = {};
    float mst[8], lst[8];
#pragma unroll
    for (int r = 0; r < 8; ++r) { mst[r] = -1e30f; lst[r] = 0.0f; }

    int buf = 0;
    for (int kb = 0; kb < kend; kb += 32, buf ^= 1) {
        wait_async0();        // this wave's copies into kbuf/vbuf[buf] are done
        __syncthreads();      // everyone's copies are visible

        if (kb + 32 < kend) issue_tiles(buf ^ 1, kb + 32);   // prefetch next block

        // K tiles from LDS in A-layout == B-layout of K^T
        const unsigned short* kp = &kbuf[buf][0];
        v16h ka0[2], ka1[2];
#pragma unroll
        for (int c = 0; c < 2; ++c) {
            HV u0, u1;
#pragma unroll
            for (int v = 0; v < 8; ++v) {
                const int kk = c * 32 + apat(v, g);
                u0.u[v] = *(const unsigned int*)(kp + (size_t)(n16)      * HDIM + kk);
                u1.u[v] = *(const unsigned int*)(kp + (size_t)(16 + n16) * HDIM + kk);
            }
            ka0[c] = u0.v; ka1[c] = u1.v;
        }

        v8f s0 = {}, s1 = {};
        s0 = __builtin_amdgcn_wmma_f32_16x16x32_f16(false, qa[0], false, ka0[0], (short)0, s0, false, false);
        s0 = __builtin_amdgcn_wmma_f32_16x16x32_f16(false, qa[1], false, ka0[1], (short)0, s0, false, false);
        s1 = __builtin_amdgcn_wmma_f32_16x16x32_f16(false, qa[0], false, ka1[0], (short)0, s1, false, false);
        s1 = __builtin_amdgcn_wmma_f32_16x16x32_f16(false, qa[1], false, ka1[1], (short)0, s1, false, false);

        const int key0 = kb + n16;
        const int key1 = kb + 16 + n16;

        float p0[8], p1[8], mloc[8];
#pragma unroll
        for (int r = 0; r < 8; ++r) {
            const int t = q0 + r + (g << 3);
            float sv0 = s0[r], sv1 = s1[r];
            const bool ok0 = (key0 <= t) && (mbase[(size_t)t * SEQ + key0] != 0);
            const bool ok1 = (key1 <= t) && (mbase[(size_t)t * SEQ + key1] != 0);
            sv0 = ok0 ? sv0 : -1e9f;
            sv1 = ok1 ? sv1 : -1e9f;
            p0[r] = sv0; p1[r] = sv1;
            mloc[r] = fmaxf(sv0, sv1);
        }
        // row-max across the 16 lanes holding each row (xor within 16-lane halves)
#pragma unroll
        for (int r = 0; r < 8; ++r) {
#pragma unroll
            for (int sh = 1; sh <= 8; sh <<= 1)
                mloc[r] = fmaxf(mloc[r], __shfl_xor(mloc[r], sh, 32));
        }

#pragma unroll
        for (int r = 0; r < 8; ++r) {
            const float mn = fmaxf(mst[r], mloc[r]);
            const float alpha = __expf(mst[r] - mn);
            mst[r] = mn;
            lst[r] *= alpha;
            o0[r] *= alpha; o1[r] *= alpha; o2[r] *= alpha; o3[r] *= alpha;
            p0[r] = __expf(p0[r] - mn);
            p1[r] = __expf(p1[r] - mn);
            float ps = p0[r] + p1[r];
#pragma unroll
            for (int sh = 1; sh <= 8; sh <<= 1)
                ps += __shfl_xor(ps, sh, 32);
            lst[r] += ps;
            const int m = r + (g << 3);
            myP[m * 32 + n16]      = f2h(p0[r]);
            myP[m * 32 + 16 + n16] = f2h(p1[r]);
        }

        __builtin_amdgcn_wave_barrier();   // LDS is in-order per wave; fence the compiler

        // reload P (16x32) in A-layout
        v16h pa;
        {
            HV u;
#pragma unroll
            for (int v = 0; v < 8; ++v)
                u.u[v] = *(volatile const unsigned int*)(myP + n16 * 32 + apat(v, g));
            pa = u.v;
        }
        __builtin_amdgcn_wave_barrier();

        // V tiles (32 keys x 16 d) from LDS V^T: contiguous key-pairs per lane
        const unsigned short* vp = &vbuf[buf][0];
#pragma unroll
        for (int aidx = 0; aidx < 4; ++aidx) {
            HV u;
#pragma unroll
            for (int v = 0; v < 8; ++v)
                u.u[v] = *(const unsigned int*)(vp + (size_t)(aidx * 16 + n16) * 32 + apat(v, g));
            v8f* op = (aidx == 0) ? &o0 : (aidx == 1) ? &o1 : (aidx == 2) ? &o2 : &o3;
            *op = __builtin_amdgcn_wmma_f32_16x16x32_f16(false, pa, false, u.v,
                                                         (short)0, *op, false, false);
        }

        __syncthreads();   // all waves done reading buf before it is overwritten again
    }

    // epilogue: normalize and store y[b][t][h*64 + d]
#pragma unroll
    for (int r = 0; r < 8; ++r) {
        const float inv = 1.0f / lst[r];
        const int t = q0 + r + (g << 3);
        float* orow = out + ((size_t)b * SEQ + t) * CDIM + h * HDIM;
        orow[0  + n16] = o0[r] * inv;
        orow[16 + n16] = o1[r] * inv;
        orow[32 + n16] = o2[r] * inv;
        orow[48 + n16] = o3[r] * inv;
    }
}

// ---------- host launcher ----------
extern "C" void kernel_launch(void* const* d_in, const int* in_sizes, int n_in,
                              void* d_out, int out_size, void* d_ws, size_t ws_size,
                              hipStream_t stream) {
    const float*         x    = (const float*)d_in[0];
    const unsigned char* mask = (const unsigned char*)d_in[1];
    const float*         W    = (const float*)d_in[2];
    const float*         bias = (const float*)d_in[3];
    float*               out  = (float*)d_out;

    unsigned short* xh  = (unsigned short*)d_ws;                 // BT*C
    unsigned short* wh  = xh + (size_t)BT * CDIM;                // 3C*C
    unsigned short* qh  = wh + (size_t)C3 * CDIM;                // B*H*T*hd
    unsigned short* kh  = qh + (size_t)BATCH * NHEAD * SEQ * HDIM;
    unsigned short* vth = kh + (size_t)BATCH * NHEAD * SEQ * HDIM;

    cvt_f32_f16<<<2048, 256, 0, stream>>>(x, xh, BT * CDIM);
    cvt_f32_f16<<<2048, 256, 0, stream>>>(W, wh, C3 * CDIM);

    // (BT/16) * (3C/64) tiles = 256*48 = 12288 waves, 8 waves/block
    qkv_gemm<<<1536, 256, 0, stream>>>(xh, wh, bias, qh, kh, vth);

    // B*H*(T/64) = 1024 blocks, 4 waves each (64 query rows per block)
    flash_attn<<<1024, 128, 0, stream>>>(qh, kh, vth, mask, out);
}
